// WinEncoderTransformer_51917564674499
// MI455X (gfx1250) — compile-verified, matrix operand order checked by reading, not run on hardware
//
#include <hip/hip_runtime.h>

typedef _Float16 f16;
typedef _Float16 v16h __attribute__((ext_vector_type(16)));
typedef _Float16 v8h  __attribute__((ext_vector_type(8)));
typedef float    v8f  __attribute__((ext_vector_type(8)));

#define T_TOK 131072        // N*B = 128*1024 tokens
#define SCALE 0.17677669529663687f   // 32^-0.5

// ---------------- helpers ----------------

__device__ __forceinline__ float wave_sum(float x) {
#pragma unroll
  for (int o = 16; o > 0; o >>= 1) x += __shfl_xor(x, o, 32);
  return x;
}

// Load one WMMA 16x32 f16 fragment (A layout; B uses same layout on W^T).
// p = pointer to this lane's row start (row-major, rows of length K), already
// offset by the current k-step. Element e of lane L maps to
// k = (e&7) + (e>>3)*16 + 8*(L>>4)  -> two contiguous 16B chunks.
__device__ __forceinline__ v16h frag_load(const f16* p, int lane) {
  const int half = (lane >> 4) & 1;
  const v8h c0 = *reinterpret_cast<const v8h*>(p + half * 8);
  const v8h c1 = *reinterpret_cast<const v8h*>(p + half * 8 + 16);
  return __builtin_shufflevector(c0, c1, 0,1,2,3,4,5,6,7,8,9,10,11,12,13,14,15);
}

// ---------------- elementwise converts ----------------

__global__ __launch_bounds__(256) void cvt_f32_f16(const float* __restrict__ in,
                                                   f16* __restrict__ out, int n) {
  int i = blockIdx.x * 256 + threadIdx.x;
  if (i < n) out[i] = (f16)in[i];
}

// Wt[n*K + k] = (f16) W[k*N + n]   (transpose + convert, W row-major [K][N])
__global__ __launch_bounds__(256) void wcvt_t(const float* __restrict__ W,
                                              f16* __restrict__ Wt, int K, int N) {
  int i = blockIdx.x * 256 + threadIdx.x;
  if (i < K * N) {
    int k = i / N, n = i % N;
    Wt[(size_t)n * K + k] = (f16)W[i];
  }
}

// ---------------- WMMA GEMM ----------------
// Y[M,N] = X[M,K] @ Wt^T  (+bias) (+relu), Wt stored [N][K] f16.
// Block = 256 threads = 8 waves; block tile = 256(M) x 64(N); wave tile 32x64:
// 2 A-frags x 4 B-frags -> 8 wmma per K-step, B frags reused across both A.
// grid.x = M/256, grid.y = N/64.  Output either f16 (stride N) or f32
// (stride ostride, column offset ocoloff).
__global__ __launch_bounds__(256) void gemm_f16w(
    const f16* __restrict__ X, const f16* __restrict__ Wt,
    const float* __restrict__ bias,
    f16* __restrict__ out16, float* __restrict__ out32,
    int K, int N, int ostride, int ocoloff, int relu) {
  const int lane = threadIdx.x & 31;
  const int wave = threadIdx.x >> 5;
  const int mrow = blockIdx.x * 256 + wave * 32;
  const int nb   = blockIdx.y * 64;
  const int ml   = lane & 15;
  const int half = lane >> 4;

  const f16* xrow0 = X  + (size_t)(mrow +      ml) * K;
  const f16* xrow1 = X  + (size_t)(mrow + 16 + ml) * K;
  const f16* w0    = Wt + (size_t)(nb +      ml) * K;
  const f16* w1    = Wt + (size_t)(nb + 16 + ml) * K;
  const f16* w2    = Wt + (size_t)(nb + 32 + ml) * K;
  const f16* w3    = Wt + (size_t)(nb + 48 + ml) * K;

  const v8f z = {0.f,0.f,0.f,0.f,0.f,0.f,0.f,0.f};
  v8f acc[2][4] = {{z, z, z, z}, {z, z, z, z}};

  for (int ks = 0; ks < K; ks += 32) {
    const v16h a0 = frag_load(xrow0 + ks, lane);
    const v16h a1 = frag_load(xrow1 + ks, lane);
    const v16h b0 = frag_load(w0 + ks, lane);
    acc[0][0] = __builtin_amdgcn_wmma_f32_16x16x32_f16(false, a0, false, b0, (short)0, acc[0][0], false, false);
    acc[1][0] = __builtin_amdgcn_wmma_f32_16x16x32_f16(false, a1, false, b0, (short)0, acc[1][0], false, false);
    const v16h b1 = frag_load(w1 + ks, lane);
    acc[0][1] = __builtin_amdgcn_wmma_f32_16x16x32_f16(false, a0, false, b1, (short)0, acc[0][1], false, false);
    acc[1][1] = __builtin_amdgcn_wmma_f32_16x16x32_f16(false, a1, false, b1, (short)0, acc[1][1], false, false);
    const v16h b2 = frag_load(w2 + ks, lane);
    acc[0][2] = __builtin_amdgcn_wmma_f32_16x16x32_f16(false, a0, false, b2, (short)0, acc[0][2], false, false);
    acc[1][2] = __builtin_amdgcn_wmma_f32_16x16x32_f16(false, a1, false, b2, (short)0, acc[1][2], false, false);
    const v16h b3 = frag_load(w3 + ks, lane);
    acc[0][3] = __builtin_amdgcn_wmma_f32_16x16x32_f16(false, a0, false, b3, (short)0, acc[0][3], false, false);
    acc[1][3] = __builtin_amdgcn_wmma_f32_16x16x32_f16(false, a1, false, b3, (short)0, acc[1][3], false, false);
  }

#pragma unroll
  for (int u = 0; u < 2; u++) {
#pragma unroll
    for (int t = 0; t < 4; t++) {
      const int n = nb + t * 16 + ml;
      const float bv = bias ? bias[n] : 0.0f;
#pragma unroll
      for (int r = 0; r < 8; r++) {
        const int row = mrow + u * 16 + half * 8 + r;
        float v = acc[u][t][r] + bv;
        if (relu) v = v > 0.f ? v : 0.f;
        if (out16) out16[(size_t)row * N + n] = (f16)v;
        else       out32[(size_t)row * ostride + ocoloff + n] = v;
      }
    }
  }
}

// ---------------- hifi: 2x2 windowed attention ----------------
// qkv layout per token row (384): q = h*32+d, k = 128+h*32+d, v = 256+h*32+d.
// One wave per (window, head); lane = head dim d. 131072 tasks.
__global__ __launch_bounds__(256) void hifi_attn(const f16* __restrict__ qkv,
                                                 f16* __restrict__ hi) {
  const int lane = threadIdx.x & 31;
  const int wave = threadIdx.x >> 5;
  const int task = blockIdx.x * 8 + wave;
  const int h   = task & 3;
  const int win = task >> 2;          // 32768 windows
  const int bc  = win >> 5;
  const int tg  = win & 31;
  const int hg  = tg >> 3;
  const int wg  = tg & 7;
  const int base = bc * 128 + hg * 32 + wg * 2;
  const int tok[4] = {base, base + 1, base + 16, base + 17};

  float q[4], k[4], v[4];
#pragma unroll
  for (int i = 0; i < 4; i++) {
    const f16* row = qkv + (size_t)tok[i] * 384 + h * 32 + lane;
    q[i] = (float)row[0];
    k[i] = (float)row[128];
    v[i] = (float)row[256];
  }
  float s[4][4];
#pragma unroll
  for (int i = 0; i < 4; i++)
#pragma unroll
    for (int j = 0; j < 4; j++)
      s[i][j] = wave_sum(q[i] * k[j]) * SCALE;

#pragma unroll
  for (int i = 0; i < 4; i++) {
    float m = s[i][0];
#pragma unroll
    for (int j = 1; j < 4; j++) m = fmaxf(m, s[i][j]);
    float e[4], sum = 0.f;
#pragma unroll
    for (int j = 0; j < 4; j++) { e[j] = __expf(s[i][j] - m); sum += e[j]; }
    const float inv = 1.0f / sum;
    float o = 0.f;
#pragma unroll
    for (int j = 0; j < 4; j++) o += e[j] * inv * v[j];
    hi[(size_t)tok[i] * 128 + h * 32 + lane] = (f16)o;
  }
}

// ---------------- lofi: window mean-pool ----------------
__global__ __launch_bounds__(256) void pool_kern(const float* __restrict__ x,
                                                 f16* __restrict__ xp) {
  const int g = blockIdx.x * 256 + threadIdx.x;   // 32768*256 exact
  const int c   = g & 255;
  const int win = g >> 8;
  const int bc = win >> 5, tg = win & 31, hg = tg >> 3, wg = tg & 7;
  const size_t base = ((size_t)bc * 128 + hg * 32 + wg * 2) * 256 + c;
  const float m = 0.25f * (x[base] + x[base + 256] + x[base + 16 * 256] + x[base + 17 * 256]);
  xp[g] = (f16)m;
}

// ---------------- lofi attention: 128 queries vs 32 pooled keys ----------------
// One block (128 thr) per (bc, head); K/V staged in LDS; one thread per token.
__global__ __launch_bounds__(128) void lofi_attn(const f16* __restrict__ q16,
                                                 const f16* __restrict__ kv16,
                                                 f16* __restrict__ lo) {
  __shared__ __align__(16) f16 kb[1024];
  __shared__ __align__(16) f16 vb[1024];
  const int bc = blockIdx.x >> 2;
  const int h  = blockIdx.x & 3;
  for (int idx = threadIdx.x; idx < 1024; idx += 128) {
    const int j = idx >> 5, d = idx & 31;
    const size_t p = ((size_t)bc * 32 + j) * 256 + h * 32 + d;
    kb[idx] = kv16[p];
    vb[idx] = kv16[p + 128];
  }
  __syncthreads();

  const int tok = bc * 128 + threadIdx.x;
  const f16* qp = q16 + (size_t)tok * 128 + h * 32;
  float q[32];
#pragma unroll
  for (int c = 0; c < 4; c++) {
    v8h qq = reinterpret_cast<const v8h*>(qp)[c];
#pragma unroll
    for (int e = 0; e < 8; e++) q[c * 8 + e] = (float)qq[e];
  }
  float s[32]; float m = -1e30f;
#pragma unroll
  for (int j = 0; j < 32; j++) {
    float acc = 0.f;
    const v8h* kr = reinterpret_cast<const v8h*>(kb + j * 32);
#pragma unroll
    for (int c = 0; c < 4; c++) {
      v8h kk = kr[c];
#pragma unroll
      for (int e = 0; e < 8; e++) acc += q[c * 8 + e] * (float)kk[e];
    }
    s[j] = acc * SCALE;
    m = fmaxf(m, s[j]);
  }
  float sum = 0.f;
#pragma unroll
  for (int j = 0; j < 32; j++) { s[j] = __expf(s[j] - m); sum += s[j]; }
  const float inv = 1.0f / sum;
  float o[32];
#pragma unroll
  for (int d = 0; d < 32; d++) o[d] = 0.f;
#pragma unroll
  for (int j = 0; j < 32; j++) {
    const float a = s[j] * inv;
    const v8h* vr = reinterpret_cast<const v8h*>(vb + j * 32);
#pragma unroll
    for (int c = 0; c < 4; c++) {
      v8h vv = vr[c];
#pragma unroll
      for (int e = 0; e < 8; e++) o[c * 8 + e] += a * (float)vv[e];
    }
  }
  f16* op = lo + (size_t)tok * 128 + h * 32;
#pragma unroll
  for (int c = 0; c < 4; c++) {
    v8h ov;
#pragma unroll
    for (int e = 0; e < 8; e++) ov[e] = (f16)o[c * 8 + e];
    reinterpret_cast<v8h*>(op)[c] = ov;
  }
}

// ---------------- residual add + LayerNorm (row=256), one wave per row ----------
__global__ __launch_bounds__(256) void add_ln(const float* __restrict__ xin,
                                              const float* __restrict__ res,
                                              const float* __restrict__ g,
                                              const float* __restrict__ b,
                                              float* __restrict__ out,
                                              f16* __restrict__ out16) {
  const int lane = threadIdx.x & 31;
  const int wave = threadIdx.x >> 5;
  const size_t row = (size_t)blockIdx.x * 8 + wave;
  const size_t off = row * 256;
  float x[8], psum = 0.f;
#pragma unroll
  for (int r = 0; r < 8; r++) {
    const int c = r * 32 + lane;
    float v = xin[off + c];
    if (res) v += res[off + c];
    x[r] = v; psum += v;
  }
  const float mean = wave_sum(psum) * (1.0f / 256.0f);
  float pv = 0.f;
#pragma unroll
  for (int r = 0; r < 8; r++) { const float d = x[r] - mean; pv += d * d; }
  const float var = wave_sum(pv) * (1.0f / 256.0f);
  const float inv = rsqrtf(var + 1e-5f);
#pragma unroll
  for (int r = 0; r < 8; r++) {
    const int c = r * 32 + lane;
    const float y = (x[r] - mean) * inv * g[c] + b[c];
    out[off + c] = y;
    if (out16) out16[off + c] = (f16)y;
  }
}

// ---------------- host orchestration ----------------

extern "C" void kernel_launch(void* const* d_in, const int* in_sizes, int n_in,
                              void* d_out, int out_size, void* d_ws, size_t ws_size,
                              hipStream_t stream) {
  (void)in_sizes; (void)n_in; (void)out_size; (void)ws_size;
  const float* tgt   = (const float*)d_in[0];
  const float* Wlq   = (const float*)d_in[1];
  const float* Wlkv  = (const float*)d_in[2];
  const float* Wlp   = (const float*)d_in[3];
  const float* blp   = (const float*)d_in[4];
  const float* Whqkv = (const float*)d_in[5];
  const float* Whp   = (const float*)d_in[6];
  const float* bhp   = (const float*)d_in[7];
  const float* W1    = (const float*)d_in[8];
  const float* b1    = (const float*)d_in[9];
  const float* W2    = (const float*)d_in[10];
  const float* b2    = (const float*)d_in[11];
  const float* ln1g  = (const float*)d_in[12];
  const float* ln1b  = (const float*)d_in[13];
  const float* ln2g  = (const float*)d_in[14];
  const float* ln2b  = (const float*)d_in[15];
  const float* lnfg  = (const float*)d_in[16];
  const float* lnfb  = (const float*)d_in[17];

  char* ws = (char*)d_ws;
  const size_t T = T_TOK;
  size_t o = 0;
  float* bufA  = (float*)(ws + o); o += T * 256 * 4;
  float* bufB  = (float*)(ws + o); o += T * 256 * 4;
  f16* x16     = (f16*)(ws + o);   o += T * 256 * 2;
  f16* big16   = (f16*)(ws + o);   o += T * 512 * 2;   // qkv (T x 384) then h1 (T x 512)
  f16* sA      = (f16*)(ws + o);   o += T * 128 * 2;   // hi_att, then lo_att
  f16* sB      = (f16*)(ws + o);   o += T * 128 * 2;   // lofi q
  f16* kv16    = (f16*)(ws + o);   o += (size_t)32768 * 256 * 2;
  f16* xp16    = (f16*)(ws + o);   o += (size_t)32768 * 256 * 2;
  f16* WhqkvT  = (f16*)(ws + o);   o += 384 * 256 * 2;
  f16* WlqT    = (f16*)(ws + o);   o += 128 * 256 * 2;
  f16* WlkvT   = (f16*)(ws + o);   o += 256 * 256 * 2;
  f16* WlpT    = (f16*)(ws + o);   o += 128 * 128 * 2;
  f16* WhpT    = (f16*)(ws + o);   o += 128 * 128 * 2;
  f16* W1T     = (f16*)(ws + o);   o += 256 * 512 * 2;
  f16* W2T     = (f16*)(ws + o);   o += 512 * 256 * 2;

  const dim3 blk(256);
  cvt_f32_f16<<<(int)((T * 256 + 255) / 256), blk, 0, stream>>>(tgt, x16, (int)(T * 256));

  for (int i = 0; i < 2; i++) {
    const float* cur = (i == 0) ? tgt : bufA;
    // weight transpose+convert (tiny)
    wcvt_t<<<(256 * 384 + 255) / 256, blk, 0, stream>>>(Whqkv + (size_t)i * 256 * 384, WhqkvT, 256, 384);
    wcvt_t<<<(256 * 128 + 255) / 256, blk, 0, stream>>>(Wlq   + (size_t)i * 256 * 128, WlqT,   256, 128);
    wcvt_t<<<(256 * 256 + 255) / 256, blk, 0, stream>>>(Wlkv  + (size_t)i * 256 * 256, WlkvT,  256, 256);
    wcvt_t<<<(128 * 128 + 255) / 256, blk, 0, stream>>>(Wlp   + (size_t)i * 128 * 128, WlpT,   128, 128);
    wcvt_t<<<(128 * 128 + 255) / 256, blk, 0, stream>>>(Whp   + (size_t)i * 128 * 128, WhpT,   128, 128);
    wcvt_t<<<(256 * 512 + 255) / 256, blk, 0, stream>>>(W1    + (size_t)i * 256 * 512, W1T,    256, 512);
    wcvt_t<<<(512 * 256 + 255) / 256, blk, 0, stream>>>(W2    + (size_t)i * 512 * 256, W2T,    512, 256);

    // hifi path
    gemm_f16w<<<dim3(512, 6), blk, 0, stream>>>(x16, WhqkvT, nullptr, big16, nullptr, 256, 384, 0, 0, 0);
    hifi_attn<<<16384, blk, 0, stream>>>(big16, sA);
    gemm_f16w<<<dim3(512, 2), blk, 0, stream>>>(sA, WhpT, bhp + (size_t)i * 128, nullptr, bufB, 128, 128, 256, 0, 0);
    // lofi path
    pool_kern<<<32768, blk, 0, stream>>>(cur, xp16);
    gemm_f16w<<<dim3(512, 2), blk, 0, stream>>>(x16, WlqT, nullptr, sB, nullptr, 256, 128, 0, 0, 0);
    gemm_f16w<<<dim3(128, 4), blk, 0, stream>>>(xp16, WlkvT, nullptr, kv16, nullptr, 256, 256, 0, 0, 0);
    lofi_attn<<<4096, dim3(128), 0, stream>>>(sB, kv16, sA);
    gemm_f16w<<<dim3(512, 2), blk, 0, stream>>>(sA, WlpT, blp + (size_t)i * 128, nullptr, bufB, 128, 128, 256, 128, 0);
    // residual + LN1 (also emits f16 for FFN input)
    add_ln<<<16384, blk, 0, stream>>>(cur, bufB, ln1g + (size_t)i * 256, ln1b + (size_t)i * 256, bufA, x16);
    // FFN
    gemm_f16w<<<dim3(512, 8), blk, 0, stream>>>(x16, W1T, b1 + (size_t)i * 512, big16, nullptr, 256, 512, 0, 0, 1);
    gemm_f16w<<<dim3(512, 4), blk, 0, stream>>>(big16, W2T, b2 + (size_t)i * 256, nullptr, bufB, 512, 256, 256, 0, 0);
    // residual + LN2 (emits f16 for next layer's GEMMs)
    add_ln<<<16384, blk, 0, stream>>>(bufA, bufB, ln2g + (size_t)i * 256, ln2b + (size_t)i * 256, bufA, x16);
  }
  // final LayerNorm into output
  add_ln<<<16384, blk, 0, stream>>>(bufA, nullptr, lnfg, lnfb, (float*)d_out, nullptr);
}
